// MultiHeadAttention_15255723835598
// MI455X (gfx1250) — matrix operand order repaired and checked by reading, not compiled, not run on hardware
//
#include <hip/hip_runtime.h>
#include <hip/hip_bf16.h>

typedef __attribute__((ext_vector_type(16))) __bf16 v16bf;
typedef __attribute__((ext_vector_type(8)))  float  v8f;

union FragBF { uint4 u4[2]; v16bf v; };

#define WMMA_BF16(a, b, c) \
  __builtin_amdgcn_wmma_f32_16x16x32_bf16(false, (a), false, (b), (short)0, (c), false, false)

// --- CDNA5 async global->LDS path (ASYNCcnt), guarded for toolchain drift ---
#if __has_builtin(__builtin_amdgcn_global_load_async_to_lds_b128)
#define HAVE_ASYNC_LDS 1
typedef int v4i __attribute__((ext_vector_type(4)));
typedef __attribute__((address_space(1))) v4i GAS4;  // global v4i
typedef __attribute__((address_space(3))) v4i LDS4;  // LDS v4i
#endif

#if __has_builtin(__builtin_amdgcn_s_wait_asynccnt)
#define WAIT_ASYNC() __builtin_amdgcn_s_wait_asynccnt(0)
#else
#define WAIT_ASYNC() asm volatile("s_wait_asynccnt 0x0" ::: "memory")
#endif

// ---------------------------------------------------------------------------
// fp32 -> bf16 conversion (n divisible by 4), packed 8B stores
// ---------------------------------------------------------------------------
__global__ void f32_to_bf16_kernel(const float* __restrict__ src,
                                   __bf16* __restrict__ dst, int n) {
  int i = (blockIdx.x * blockDim.x + threadIdx.x) * 4;
  if (i < n) {
    float4 f = *(const float4*)(src + i);
    union { __bf16 h[4]; uint2 u; } o;
    o.h[0] = (__bf16)f.x; o.h[1] = (__bf16)f.y;
    o.h[2] = (__bf16)f.z; o.h[3] = (__bf16)f.w;
    *(uint2*)(dst + i) = o.u;
  }
}

// ---------------------------------------------------------------------------
// Tiled bf16 WMMA GEMM: C[M,N] = A[M,K] @ B[K,N] + bias
// Block: 256 threads = 8 waves; tile 128(M) x 128(N), K-step 32.
// Wave grid 4x2 -> each wave computes 32x64 (2x4 WMMA tiles, 8 WMMA/step).
// A tile staged with GLOBAL_LOAD_ASYNC_TO_LDS_B128 when available.
// MODE 0: write bf16 to q layout [B*H, L, 64]
// MODE 1: cols <1024 -> k layout [B*H, L, 64]; cols >=1024 -> vT [B*H, 64, L]
// MODE 2: write fp32 row-major (final output)
// ---------------------------------------------------------------------------
template <int MODE>
__global__ __launch_bounds__(256) void gemm_bf16_wmma(
    const __bf16* __restrict__ A, const __bf16* __restrict__ Bw,
    const float* __restrict__ bias, int M, int N, int K,
    __bf16* __restrict__ outQ, __bf16* __restrict__ outV,
    float* __restrict__ outF) {
  __shared__ __align__(16) __bf16 As[128 * 32];
  __shared__ __align__(16) __bf16 Bs[128 * 32];  // transposed: [n][k]

  const int tid  = threadIdx.x;
  const int lane = tid & 31;
  const int w    = tid >> 5;
  const int wm   = w & 3;   // 0..3 -> M sub-tile
  const int wn   = w >> 2;  // 0..1 -> N sub-tile
  const int mBase = blockIdx.y * 128;
  const int nBase = blockIdx.x * 128;

  const int l15   = lane & 15;
  const int lh    = (lane & 16) ? 8 : 0;   // A-frag k-half offset
  const int bkoff = (lane & 16) ? 8 : 0;   // B-frag k offset

  const int aRow = tid >> 1;         // 0..127
  const int aK   = (tid & 1) * 16;   // 0 or 16
  const int bKr  = tid >> 3;         // 0..31
  const int bN0  = (tid & 7) * 16;   // 0..112

  v8f acc[2][4];
  #pragma unroll
  for (int i = 0; i < 2; i++)
    #pragma unroll
    for (int j = 0; j < 4; j++) acc[i][j] = {};

  for (int k0 = 0; k0 < K; k0 += 32) {
    // stage A tile 128x32 (each thread: one 32B chunk)
    {
      const __bf16* gp = A + (size_t)(mBase + aRow) * K + k0 + aK;
#if defined(HAVE_ASYNC_LDS)
      // direct global->LDS DMA, no VGPR round-trip (ASYNCcnt tracked)
      __builtin_amdgcn_global_load_async_to_lds_b128(
          (GAS4*)gp, (LDS4*)&As[aRow * 32 + aK], 0, 0);
      __builtin_amdgcn_global_load_async_to_lds_b128(
          (GAS4*)(gp + 8), (LDS4*)&As[aRow * 32 + aK + 8], 0, 0);
#else
      uint4 t0 = *(const uint4*)gp;
      uint4 t1 = *(const uint4*)(gp + 8);
      *(uint4*)(&As[aRow * 32 + aK])     = t0;
      *(uint4*)(&As[aRow * 32 + aK + 8]) = t1;
#endif
      if (k0 + 32 < K) __builtin_prefetch(gp + 32, 0, 1);
    }
    // stage B tile 32x128, transposed into Bs[n][k] (register transpose)
    {
      const __bf16* gp = Bw + (size_t)(k0 + bKr) * N + nBase + bN0;
      uint4 t0 = *(const uint4*)gp;
      uint4 t1 = *(const uint4*)(gp + 8);
      __bf16 tmp[16];
      *(uint4*)tmp       = t0;
      *(uint4*)(tmp + 8) = t1;
      #pragma unroll
      for (int j = 0; j < 16; j++) Bs[(bN0 + j) * 32 + bKr] = tmp[j];
    }
#if defined(HAVE_ASYNC_LDS)
    WAIT_ASYNC();
#endif
    __syncthreads();

    #pragma unroll
    for (int i = 0; i < 2; i++) {
      FragBF a;
      const __bf16* ap = &As[(wm * 32 + i * 16 + l15) * 32];
      a.u4[0] = *(const uint4*)(ap + lh);
      a.u4[1] = *(const uint4*)(ap + 16 + lh);
      #pragma unroll
      for (int j = 0; j < 4; j++) {
        FragBF b;
        const __bf16* bp = &Bs[(wn * 64 + j * 16 + l15) * 32 + bkoff];
        b.u4[0] = *(const uint4*)(bp);
        b.u4[1] = *(const uint4*)(bp + 16);
        acc[i][j] = WMMA_BF16(a.v, b.v, acc[i][j]);
      }
    }
    __syncthreads();
  }

  // epilogue
  #pragma unroll
  for (int i = 0; i < 2; i++) {
    #pragma unroll
    for (int j = 0; j < 4; j++) {
      const int col = nBase + wn * 64 + j * 16 + l15;
      const float bv = bias[col];
      #pragma unroll
      for (int r = 0; r < 8; r++) {
        const int row = mBase + wm * 32 + i * 16 + r + ((lane & 16) ? 8 : 0);
        const float val = acc[i][j][r] + bv;
        if (MODE == 2) {
          outF[(size_t)row * N + col] = val;
        } else if (MODE == 0) {
          const int b = row >> 11, l = row & 2047;
          const int h = col >> 6, c = col & 63;
          outQ[(((size_t)(b * 16 + h)) * 2048 + l) * 64 + c] = (__bf16)val;
        } else {  // MODE 1: KV
          const int b = row >> 11, l = row & 2047;
          if (col < 1024) {
            const int h = col >> 6, c = col & 63;
            outQ[(((size_t)(b * 16 + h)) * 2048 + l) * 64 + c] = (__bf16)val;
          } else {
            const int cv = col - 1024;
            const int h = cv >> 6, c = cv & 63;
            outV[(((size_t)(b * 16 + h)) * 64 + c) * 2048 + l] = (__bf16)val;
          }
        }
      }
    }
  }
}

// ---------------------------------------------------------------------------
// Flash attention: grid (L/64, B*H), block 128 threads = 4 waves.
// Each wave: 16 query rows, streams 64-key tiles with online softmax
// (16 WMMAs per tile; softmax bookkeeping amortized over 64 keys).
// q,k: [B*H, L, 64] bf16 ; vt: [B*H, 64, L] bf16 ; y: [B, L, 1024] bf16
// Softmax scale (1/8, exact power of two) is folded into Q at load time.
// ---------------------------------------------------------------------------
__global__ __launch_bounds__(128) void flash_attn_wmma(
    const __bf16* __restrict__ q, const __bf16* __restrict__ k,
    const __bf16* __restrict__ vt, __bf16* __restrict__ y) {
  __shared__ __align__(16) __bf16 Plds[4][16 * 64];

  const int lane = threadIdx.x & 31;
  const int w    = threadIdx.x >> 5;  // 0..3
  const int l15  = lane & 15;
  const int hi   = (lane & 16) ? 1 : 0;
  const int bh   = blockIdx.y;                 // 0..31
  const int q0   = blockIdx.x * 64 + w * 16;   // query row base for this wave
  const float L2E = 1.4426950408889634f;

  // load Q A-fragments (c = 0..31 and 32..63) and fold in the 1/8 scale
  FragBF aQ[2];
  {
    const __bf16* qp = q + ((size_t)bh * 2048 + q0 + l15) * 64;
    const int lh = hi * 8;
    aQ[0].u4[0] = *(const uint4*)(qp + lh);
    aQ[0].u4[1] = *(const uint4*)(qp + 16 + lh);
    aQ[1].u4[0] = *(const uint4*)(qp + 32 + lh);
    aQ[1].u4[1] = *(const uint4*)(qp + 48 + lh);
    #pragma unroll
    for (int e = 0; e < 16; e++) {
      aQ[0].v[e] = (__bf16)((float)aQ[0].v[e] * 0.125f);  // exact in bf16
      aQ[1].v[e] = (__bf16)((float)aQ[1].v[e] * 0.125f);
    }
  }

  v8f O[4];
  #pragma unroll
  for (int i = 0; i < 4; i++) O[i] = {};
  float m[8], lsum[8];
  #pragma unroll
  for (int r = 0; r < 8; r++) { m[r] = -3.0e38f; lsum[r] = 0.f; }

  const __bf16* kbase = k  + (size_t)bh * 2048 * 64;
  const __bf16* vbase = vt + (size_t)bh * 64 * 2048;
  __bf16* pl = Plds[w];  // wave-private 16x64 patch

  for (int kt = 0; kt < 2048; kt += 64) {
    // ---- S = (Q @ K^T): 16 queries x 64 keys, acc over HD=64
    v8f s[4];
    #pragma unroll
    for (int f = 0; f < 4; f++) {  // key quarter: keys kt + f*16 + [0,16)
      FragBF b0, b1;
      const __bf16* kp = kbase + ((size_t)(kt + f * 16 + l15)) * 64 + hi * 8;
      b0.u4[0] = *(const uint4*)(kp);
      b0.u4[1] = *(const uint4*)(kp + 16);
      b1.u4[0] = *(const uint4*)(kp + 32);
      b1.u4[1] = *(const uint4*)(kp + 48);
      v8f c = {};
      c = WMMA_BF16(aQ[0].v, b0.v, c);
      c = WMMA_BF16(aQ[1].v, b1.v, c);
      s[f] = c;
    }

    // ---- online softmax (row stats across the 16-lane column group)
    #pragma unroll
    for (int r = 0; r < 8; r++) {
      const float s0 = s[0][r], s1 = s[1][r], s2 = s[2][r], s3 = s[3][r];
      float mx = fmaxf(fmaxf(s0, s1), fmaxf(s2, s3));
      #pragma unroll
      for (int msk = 8; msk >= 1; msk >>= 1)
        mx = fmaxf(mx, __shfl_xor(mx, msk, 32));
      const float mnew = fmaxf(m[r], mx);
      const float corr = exp2f((m[r] - mnew) * L2E);
      const float e0 = exp2f((s0 - mnew) * L2E);
      const float e1 = exp2f((s1 - mnew) * L2E);
      const float e2 = exp2f((s2 - mnew) * L2E);
      const float e3 = exp2f((s3 - mnew) * L2E);
      float rs = (e0 + e1) + (e2 + e3);
      #pragma unroll
      for (int msk = 8; msk >= 1; msk >>= 1)
        rs += __shfl_xor(rs, msk, 32);
      lsum[r] = lsum[r] * corr + rs;
      m[r] = mnew;
      #pragma unroll
      for (int i = 0; i < 4; i++) O[i][r] *= corr;
      // D-frag -> LDS (A-frag reshape source), row-major [16][64]
      const int ro = (r + hi * 8) * 64;
      pl[ro + l15]      = (__bf16)e0;
      pl[ro + 16 + l15] = (__bf16)e1;
      pl[ro + 32 + l15] = (__bf16)e2;
      pl[ro + 48 + l15] = (__bf16)e3;
    }

    // ---- read P back as two A-fragments (keys 0..31 and 32..63)
    FragBF aP[2];
    {
      const __bf16* pp = pl + l15 * 64;
      aP[0].u4[0] = *(const uint4*)(pp + hi * 8);
      aP[0].u4[1] = *(const uint4*)(pp + 16 + hi * 8);
      aP[1].u4[0] = *(const uint4*)(pp + 32 + hi * 8);
      aP[1].u4[1] = *(const uint4*)(pp + 48 + hi * 8);
    }

    // ---- O += P @ V  (V^T layout makes B-frags contiguous)
    #pragma unroll
    for (int nb = 0; nb < 4; nb++) {
      const __bf16* vp = vbase + ((size_t)(nb * 16 + l15)) * 2048 + kt + hi * 8;
      FragBF bV0, bV1;
      bV0.u4[0] = *(const uint4*)(vp);
      bV0.u4[1] = *(const uint4*)(vp + 16);
      bV1.u4[0] = *(const uint4*)(vp + 32);
      bV1.u4[1] = *(const uint4*)(vp + 48);
      O[nb] = WMMA_BF16(aP[0].v, bV0.v, O[nb]);
      O[nb] = WMMA_BF16(aP[1].v, bV1.v, O[nb]);
    }
  }

  // ---- normalize + store to y [B, L, H*64]
  const int b = bh >> 4, h = bh & 15;
  #pragma unroll
  for (int r = 0; r < 8; r++) {
    const float inv = 1.0f / lsum[r];
    const int lq = q0 + r + hi * 8;
    const size_t rowoff = ((size_t)b * 2048 + lq) * 1024 + h * 64;
    #pragma unroll
    for (int nb = 0; nb < 4; nb++)
      y[rowoff + nb * 16 + l15] = (__bf16)(O[nb][r] * inv);
  }
}

// ---------------------------------------------------------------------------
// Host launcher
// ---------------------------------------------------------------------------
extern "C" void kernel_launch(void* const* d_in, const int* in_sizes, int n_in,
                              void* d_out, int out_size, void* d_ws, size_t ws_size,
                              hipStream_t stream) {
  const float* x   = (const float*)d_in[0];
  const float* Wq  = (const float*)d_in[1];
  const float* bq  = (const float*)d_in[2];
  const float* Wkv = (const float*)d_in[3];
  const float* bkv = (const float*)d_in[4];
  const float* Wo  = (const float*)d_in[5];
  const float* bo  = (const float*)d_in[6];
  float* out = (float*)d_out;

  char* ws = (char*)d_ws;
  auto alloc = [&](size_t elems) -> __bf16* {
    __bf16* p = (__bf16*)ws;
    ws += ((elems * sizeof(__bf16) + 255) & ~(size_t)255);
    return p;
  };
  __bf16* xb   = alloc((size_t)4096 * 1024);
  __bf16* wqb  = alloc((size_t)1024 * 1024);
  __bf16* wkvb = alloc((size_t)1024 * 2048);
  __bf16* wob  = alloc((size_t)1024 * 1024);
  __bf16* qb   = alloc((size_t)32 * 2048 * 64);
  __bf16* kb   = alloc((size_t)32 * 2048 * 64);
  __bf16* vtb  = alloc((size_t)32 * 64 * 2048);
  __bf16* yb   = alloc((size_t)4096 * 1024);

  auto cvt = [&](const float* s, __bf16* d, int n) {
    f32_to_bf16_kernel<<<(n / 4 + 255) / 256, 256, 0, stream>>>(s, d, n);
  };
  cvt(x,   xb,   4096 * 1024);
  cvt(Wq,  wqb,  1024 * 1024);
  cvt(Wkv, wkvb, 1024 * 2048);
  cvt(Wo,  wob,  1024 * 1024);

  dim3 blk(256);
  // Q projection
  gemm_bf16_wmma<0><<<dim3(1024 / 128, 4096 / 128), blk, 0, stream>>>(
      xb, wqb, bq, 4096, 1024, 1024, qb, nullptr, nullptr);
  // KV projection (K row-major per head, V transposed per head)
  gemm_bf16_wmma<1><<<dim3(2048 / 128, 4096 / 128), blk, 0, stream>>>(
      xb, wkvb, bkv, 4096, 2048, 1024, kb, vtb, nullptr);
  // attention
  flash_attn_wmma<<<dim3(2048 / 64, 32), dim3(128), 0, stream>>>(qb, kb, vtb, yb);
  // output projection -> fp32 d_out
  gemm_bf16_wmma<2><<<dim3(1024 / 128, 4096 / 128), blk, 0, stream>>>(
      yb, wob, bo, 4096, 1024, 1024, nullptr, nullptr, out);
}
